// ObjectWordGAT_42734924595326
// MI455X (gfx1250) — matrix operand order repaired and checked by reading, not compiled, not run on hardware
//
#include <hip/hip_runtime.h>
#include <hip/hip_bf16.h>

#define D_DIM 512
#define H_DIM 2
#define E_DIM 512
#define HE_DIM 1024
#define B_DIM 256
#define NO 36
#define NW 256
#define NEG_SLOPE 0.2f

typedef __attribute__((ext_vector_type(16))) __bf16 v16bf;
typedef __attribute__((ext_vector_type(8)))  float  v8f;

union Frag16 { unsigned u[8]; v16bf v; };

__device__ __forceinline__ unsigned short f2bf(float f) {
  unsigned u = __float_as_uint(f);
  u += 0x7FFFu + ((u >> 16) & 1u);   // round-to-nearest-even
  return (unsigned short)(u >> 16);
}
__device__ __forceinline__ float bf2f(unsigned short h) {
  return __uint_as_float(((unsigned)h) << 16);
}
__device__ __forceinline__ float leaky(float z) {
  return z >= 0.f ? z : NEG_SLOPE * z;
}

// ---- K0: elementwise fp32 -> bf16 (for embeddings; single streaming pass)
__global__ void k_convert(const float* __restrict__ src, unsigned short* __restrict__ dst, int n) {
  int idx = blockIdx.x * blockDim.x + threadIdx.x;
  if (idx < n) dst[idx] = f2bf(src[idx]);
}

// ---- K1: W [D][HE] fp32 -> Wt [HE][D] bf16 (transposed so WMMA B-frags are contiguous)
__global__ void k_convert_W(const float* __restrict__ W, unsigned short* __restrict__ Wt) {
  int idx = blockIdx.x * blockDim.x + threadIdx.x;
  if (idx >= D_DIM * HE_DIM) return;
  int d = idx / HE_DIM, he = idx % HE_DIM;
  Wt[he * D_DIM + d] = f2bf(W[idx]);
}

// ---- K2: h[b,h,m,e] = X[b,m,:] @ W  via v_wmma_f32_16x16x32_bf16.
// One wave computes a 16 x (16*NT) strip: A-frag loaded once per k-step and
// reused across NT B-tiles (10 dword loads per WMMA instead of 16).
// Layouts per CDNA5 ISA 7.12.2. GUARD=false for tile-aligned M (no branches).
template<int NT, bool GUARD>
__global__ void k_gemm_h(const unsigned short* __restrict__ Xbf, int M,
                         const unsigned short* __restrict__ Wt,
                         unsigned short* __restrict__ Hout) {
  const int lane = threadIdx.x & 31;
  const int lm = lane & 15, lh = lane >> 4;
  const int heBase = blockIdx.x * (16 * NT);
  const int mBase  = blockIdx.y * 16;
  const int b      = blockIdx.z;
  int mA = mBase + lm;
  if (GUARD && mA >= M) mA = 0;                 // clamp; rows discarded at store
  const unsigned* Arow = (const unsigned*)(Xbf + ((size_t)b * M + mA) * D_DIM);
  const unsigned* Brow[NT];
#pragma unroll
  for (int t = 0; t < NT; ++t)
    Brow[t] = (const unsigned*)(Wt + (size_t)(heBase + t * 16 + lm) * D_DIM);

  v8f acc[NT] = {};
  for (int kb = 0; kb < D_DIM; kb += 32) {
    Frag16 a;
#pragma unroll
    for (int j = 0; j < 8; ++j) {
      // A layout: lanes0-15: V0..3 K=0..7, V4..7 K=16..23; lanes16-31: +8
      int kA = kb + ((j >> 2) << 4) + (lh << 3) + ((j & 3) << 1);
      a.u[j] = Arow[kA >> 1];
    }
#pragma unroll
    for (int t = 0; t < NT; ++t) {
      Frag16 bb;
#pragma unroll
      for (int j = 0; j < 8; ++j) {
        // B layout: lanes0-15: V0..3 K=0..7, V4..7 K=16..23; lanes16-31: +8
        int kB = kb + ((j >> 2) << 4) + (lh << 3) + ((j & 3) << 1);
        bb.u[j] = Brow[t][kB >> 1];
      }
      acc[t] = __builtin_amdgcn_wmma_f32_16x16x32_bf16(false, a.v, false, bb.v,
                                                       (short)0, acc[t], false, false);
    }
  }
#pragma unroll
  for (int t = 0; t < NT; ++t) {
    int nB = heBase + t * 16 + lm;
    int head = nB >> 9, e = nB & 511;
#pragma unroll
    for (int j = 0; j < 8; ++j) {              // C: VGPR j -> row m=j (+8 for hi lanes)
      int m = mBase + j + (lh << 3);
      if (!GUARD || m < M)
        Hout[(((size_t)b * H_DIM + head) * M + m) * E_DIM + e] = f2bf(acc[t][j]);
    }
  }
}

// ---- K3: per-row dots with att_src/att_dst; one wave32 per row, shfl_xor reduce
template<bool WITH_DST>
__global__ void k_scores(const unsigned short* __restrict__ Hmat, int M,
                         const float* __restrict__ att_src,
                         const float* __restrict__ att_dst,
                         float* __restrict__ s_src, float* __restrict__ s_dst) {
  int wave = blockIdx.x * (blockDim.x >> 5) + (threadIdx.x >> 5);
  int lane = threadIdx.x & 31;
  if (wave >= B_DIM * H_DIM * M) return;
  int bh = wave / M;                  // = b*H + h
  int h  = bh & 1;
  const unsigned short* hrow = Hmat + (size_t)wave * E_DIM;
  const float* asrc = att_src + h * E_DIM;
  const float* adst = att_dst + h * E_DIM;
  float accs = 0.f, accd = 0.f;
  for (int t = 0; t < E_DIM / 32; ++t) {
    int e = t * 32 + lane;
    float hv = bf2f(hrow[e]);
    accs += hv * asrc[e];
    if (WITH_DST) accd += hv * adst[e];
  }
  for (int m = 16; m > 0; m >>= 1) {
    accs += __shfl_xor(accs, m, 32);
    if (WITH_DST) accd += __shfl_xor(accd, m, 32);
  }
  if (lane == 0) {
    s_src[wave] = accs;
    if (WITH_DST) s_dst[wave] = accd;
  }
}

// ---- K4: per (b,w): softmax over 37 logits, msg, mean over heads, +bias -> uw bf16
__global__ void k_msg_updword(const unsigned short* __restrict__ h_obj,
                              const unsigned short* __restrict__ h_word,
                              const float* __restrict__ s_src_obj,
                              const float* __restrict__ s_src_word,
                              const float* __restrict__ s_dst_word,
                              const float* __restrict__ bias,
                              unsigned short* __restrict__ uw) {
  const int w = blockIdx.x, b = blockIdx.y, tid = threadIdx.x; // 256 threads
  __shared__ float alpha[NO + 1];
  __shared__ float sobj[NO];
  float acc0 = 0.f, acc1 = 0.f;
  for (int h = 0; h < H_DIM; ++h) {
    int bh = b * H_DIM + h;
    if (tid < NO) sobj[tid] = s_src_obj[bh * NO + tid];
    __syncthreads();
    float sdw = s_dst_word[bh * NW + w];
    float ssw = s_src_word[bh * NW + w];
    if (tid <= NO) {
      float mx = leaky(ssw + sdw);
      for (int n = 0; n < NO; ++n) mx = fmaxf(mx, leaky(sobj[n] + sdw));
      float sm = __expf(leaky(ssw + sdw) - mx);
      for (int n = 0; n < NO; ++n) sm += __expf(leaky(sobj[n] + sdw) - mx);
      float lg = (tid < NO) ? leaky(sobj[tid] + sdw) : leaky(ssw + sdw);
      alpha[tid] = __expf(lg - mx) / sm;
    }
    __syncthreads();
    const unsigned short* hob = h_obj + (size_t)bh * NO * E_DIM;
    const unsigned short* hwr = h_word + ((size_t)bh * NW + w) * E_DIM;
    float aself = alpha[NO];
    int e0 = tid, e1 = tid + 256;
    float m0 = aself * bf2f(hwr[e0]);
    float m1 = aself * bf2f(hwr[e1]);
    for (int n = 0; n < NO; ++n) {
      float an = alpha[n];
      m0 += an * bf2f(hob[n * E_DIM + e0]);
      m1 += an * bf2f(hob[n * E_DIM + e1]);
    }
    acc0 += m0; acc1 += m1;
    __syncthreads();
  }
  size_t base = ((size_t)b * NW + w) * E_DIM;
  uw[base + tid]       = f2bf(0.5f * acc0 + bias[tid]);
  uw[base + tid + 256] = f2bf(0.5f * acc1 + bias[tid + 256]);
}

// ---- K5: upd_obj = mean over heads of h_obj + bias -> uo bf16
__global__ void k_updobj(const unsigned short* __restrict__ h_obj,
                         const float* __restrict__ bias,
                         unsigned short* __restrict__ uo) {
  size_t idx = (size_t)blockIdx.x * blockDim.x + threadIdx.x;
  if (idx >= (size_t)B_DIM * NO * E_DIM) return;
  int e = (int)(idx & 511);
  size_t bn = idx >> 9;
  size_t b = bn / NO, n = bn % NO;
  float v0 = bf2f(h_obj[((b * 2 + 0) * NO + n) * E_DIM + e]);
  float v1 = bf2f(h_obj[((b * 2 + 1) * NO + n) * E_DIM + e]);
  uo[idx] = f2bf(0.5f * (v0 + v1) + bias[e]);
}

// ---- K6: S[b,n,w] = uo[b,n,:] . uw[b,w,:] via WMMA (K=512); NT w-tiles per wave
template<int NT>
__global__ void k_attn_logits(const unsigned short* __restrict__ uo,
                              const unsigned short* __restrict__ uw,
                              float* __restrict__ S) {
  const int lane = threadIdx.x & 31;
  const int lm = lane & 15, lh = lane >> 4;
  const int wBase = blockIdx.x * (16 * NT);
  const int mBase = blockIdx.y * 16;
  const int b = blockIdx.z;
  int mA = mBase + lm; if (mA >= NO) mA = 0;
  const unsigned* Arow = (const unsigned*)(uo + ((size_t)b * NO + mA) * E_DIM);
  const unsigned* Brow[NT];
#pragma unroll
  for (int t = 0; t < NT; ++t)
    Brow[t] = (const unsigned*)(uw + ((size_t)b * NW + (wBase + t * 16 + lm)) * E_DIM);

  v8f acc[NT] = {};
  for (int kb = 0; kb < E_DIM; kb += 32) {
    Frag16 a;
#pragma unroll
    for (int j = 0; j < 8; ++j) {
      int kA = kb + ((j >> 2) << 4) + (lh << 3) + ((j & 3) << 1);
      a.u[j] = Arow[kA >> 1];
    }
#pragma unroll
    for (int t = 0; t < NT; ++t) {
      Frag16 bb;
#pragma unroll
      for (int j = 0; j < 8; ++j) {
        int kB = kb + ((j >> 2) << 4) + (lh << 3) + ((j & 3) << 1);
        bb.u[j] = Brow[t][kB >> 1];
      }
      acc[t] = __builtin_amdgcn_wmma_f32_16x16x32_bf16(false, a.v, false, bb.v,
                                                       (short)0, acc[t], false, false);
    }
  }
#pragma unroll
  for (int t = 0; t < NT; ++t) {
    int n = wBase + t * 16 + lm;
#pragma unroll
    for (int j = 0; j < 8; ++j) {
      int m = mBase + j + (lh << 3);
      if (m < NO)
        S[((size_t)b * NO + m) * NW + n] = acc[t][j];
    }
  }
}

// ---- K7: softmax rows of S over w, fold mean_n into c[w], out = c @ uw
__global__ void k_final(const float* __restrict__ S,
                        const unsigned short* __restrict__ uw,
                        float* __restrict__ out) {
  const int b = blockIdx.x, tid = threadIdx.x;   // 256 threads, thread==w
  __shared__ float red[256];
  __shared__ float cw[256];
  float c = 0.f;
  for (int n = 0; n < NO; ++n) {
    float x = S[((size_t)b * NO + n) * NW + tid];
    red[tid] = x; __syncthreads();
    for (int s = 128; s > 0; s >>= 1) { if (tid < s) red[tid] = fmaxf(red[tid], red[tid + s]); __syncthreads(); }
    float mx = red[0]; __syncthreads();
    float ex = __expf(x - mx);
    red[tid] = ex; __syncthreads();
    for (int s = 128; s > 0; s >>= 1) { if (tid < s) red[tid] += red[tid + s]; __syncthreads(); }
    float sm = red[0]; __syncthreads();
    c += ex / sm;
  }
  cw[tid] = c * (1.f / NO);
  __syncthreads();
  for (int ee = 0; ee < 2; ++ee) {
    int e = tid + (ee << 8);
    float acc = 0.f;
    for (int w = 0; w < NW; ++w)
      acc += cw[w] * bf2f(uw[((size_t)b * NW + w) * E_DIM + e]);
    out[(size_t)b * E_DIM + e] = acc;
  }
}

extern "C" void kernel_launch(void* const* d_in, const int* in_sizes, int n_in,
                              void* d_out, int out_size, void* d_ws, size_t ws_size,
                              hipStream_t stream) {
  const float* obj  = (const float*)d_in[0];
  const float* word = (const float*)d_in[1];
  const float* W    = (const float*)d_in[2];
  const float* asrc = (const float*)d_in[3];
  const float* adst = (const float*)d_in[4];
  const float* bias = (const float*)d_in[5];
  float* out = (float*)d_out;

  char* ws = (char*)d_ws;
  size_t off = 0;
  unsigned short* Wt      = (unsigned short*)(ws + off); off += (size_t)HE_DIM * D_DIM * 2;
  unsigned short* obj_bf  = (unsigned short*)(ws + off); off += (size_t)B_DIM * NO * D_DIM * 2;
  unsigned short* word_bf = (unsigned short*)(ws + off); off += (size_t)B_DIM * NW * D_DIM * 2;
  unsigned short* h_obj   = (unsigned short*)(ws + off); off += (size_t)B_DIM * H_DIM * NO * E_DIM * 2;
  unsigned short* h_word  = (unsigned short*)(ws + off); off += (size_t)B_DIM * H_DIM * NW * E_DIM * 2;
  float* s_src_obj  = (float*)(ws + off); off += (size_t)B_DIM * H_DIM * NO * 4;
  float* s_src_word = (float*)(ws + off); off += (size_t)B_DIM * H_DIM * NW * 4;
  float* s_dst_word = (float*)(ws + off); off += (size_t)B_DIM * H_DIM * NW * 4;
  unsigned short* uw = (unsigned short*)(ws + off); off += (size_t)B_DIM * NW * E_DIM * 2;
  unsigned short* uo = (unsigned short*)(ws + off); off += (size_t)B_DIM * NO * E_DIM * 2;
  float* S = (float*)(ws + off); off += (size_t)B_DIM * NO * NW * 4;
  (void)ws_size; (void)in_sizes; (void)n_in; (void)out_size;

  const int n_obj  = B_DIM * NO * D_DIM;
  const int n_word = B_DIM * NW * D_DIM;
  k_convert<<<(n_obj + 255) / 256, 256, 0, stream>>>(obj, obj_bf, n_obj);
  k_convert<<<(n_word + 255) / 256, 256, 0, stream>>>(word, word_bf, n_word);
  k_convert_W<<<(D_DIM * HE_DIM + 255) / 256, 256, 0, stream>>>(W, Wt);

  k_gemm_h<4, false><<<dim3(HE_DIM / 64, NW / 16, B_DIM), 32, 0, stream>>>(word_bf, NW, Wt, h_word);
  k_gemm_h<4, true ><<<dim3(HE_DIM / 64, (NO + 15) / 16, B_DIM), 32, 0, stream>>>(obj_bf, NO, Wt, h_obj);

  k_scores<true ><<<(B_DIM * H_DIM * NW + 7) / 8, 256, 0, stream>>>(h_word, NW, asrc, adst, s_src_word, s_dst_word);
  k_scores<false><<<(B_DIM * H_DIM * NO + 7) / 8, 256, 0, stream>>>(h_obj, NO, asrc, adst, s_src_obj, nullptr);

  k_msg_updword<<<dim3(NW, B_DIM), 256, 0, stream>>>(h_obj, h_word, s_src_obj, s_src_word, s_dst_word, bias, uw);
  k_updobj<<<(int)(((size_t)B_DIM * NO * E_DIM + 255) / 256), 256, 0, stream>>>(h_obj, bias, uo);

  k_attn_logits<4><<<dim3(NW / 64, (NO + 15) / 16, B_DIM), 32, 0, stream>>>(uo, uw, S);
  k_final<<<B_DIM, 256, 0, stream>>>(S, uw, out);
}
